// CausalSelfAttention_6442450944521
// MI455X (gfx1250) — compile-verified, matrix operand order checked by reading, not run on hardware
//
#include <hip/hip_runtime.h>
#include <stdint.h>

// ---------------------------------------------------------------------------
// Types for WMMA fragments (CDNA5 gfx1250, wave32)
// ---------------------------------------------------------------------------
typedef __attribute__((ext_vector_type(16))) __bf16     bf16x16;
typedef __attribute__((ext_vector_type(8)))  float      f32x8;
typedef __attribute__((ext_vector_type(4)))  unsigned int u32x4;

union FragB {
    bf16x16 v;
    u32x4   q[2];   // two 16B chunks: K = 8*half .. 8*half+7 and K = 16+8*half ..
};

__device__ __forceinline__ unsigned short f32_to_bf16(float f) {
    union { float f; unsigned int u; } x;
    x.f = f;
    unsigned int u = x.u;
    return (unsigned short)((u + 0x7FFFu + ((u >> 16) & 1u)) >> 16);
}

// Async DMA: global -> LDS, 16 bytes per lane, tracked by ASYNCcnt.
__device__ __forceinline__ void async_ld_b128(const void* gaddr, void* lds_ptr) {
    unsigned loff = (unsigned)(size_t)lds_ptr;   // low 32 bits = LDS offset
    asm volatile("global_load_async_to_lds_b128 %0, %1, off"
                 :: "v"(loff), "v"(gaddr) : "memory");
}
__device__ __forceinline__ void async_wait0() {
    asm volatile("s_wait_asynccnt 0x0" ::: "memory");
}

// ---------------------------------------------------------------------------
// Problem constants
// ---------------------------------------------------------------------------
static constexpr int BATCH   = 4;
static constexpr int T_TOT   = 2048;
static constexpr int EMB     = 1024;
static constexpr int HEADS   = 16;
static constexpr int HDIM    = 64;
static constexpr int SEG     = 512;

// ---------------------------------------------------------------------------
// Elementwise convert f32 -> bf16
// ---------------------------------------------------------------------------
__global__ void cvt_f32_bf16(const float* __restrict__ in,
                             unsigned short* __restrict__ out, int n) {
    int i = blockIdx.x * blockDim.x + threadIdx.x;
    if (i < n) out[i] = f32_to_bf16(in[i]);
}

// Transpose [rows][cols] f32 -> [cols][rows] bf16
__global__ void transpose_f32_bf16(const float* __restrict__ in,
                                   unsigned short* __restrict__ out,
                                   int rows, int cols) {
    int i = blockIdx.x * blockDim.x + threadIdx.x;
    if (i < rows * cols) {
        int r = i / cols, c = i % cols;
        out[(size_t)c * rows + r] = f32_to_bf16(in[i]);
    }
}

// ---------------------------------------------------------------------------
// Tiled bf16 GEMM:  C[m][n] = sum_k A[m][k] * Bt[n][k]  (+ bias[n])
// Block = 256 threads = 8 waves (4 M-quadrants x 2 N-halves).
// Tile 128x128, K-stage 64 (two WMMA K-steps), double-buffered LDS fed by
// GLOBAL_LOAD_ASYNC_TO_LDS_B128 so DMA of stage t+1 overlaps the 16 WMMAs
// of stage t. 72 KB LDS total.
// MODE 0: scatter epilogue into q / k / v^T bf16 buffers (QKV fusion).
// MODE 1: f32 epilogue to out (output projection).
// ---------------------------------------------------------------------------
template <int MODE>
__global__ __launch_bounds__(256)
void gemm_bf16(const unsigned short* __restrict__ A,    // [M][K] bf16
               const unsigned short* __restrict__ Bt,   // [N][K] bf16
               const float* __restrict__ bias,          // [N]
               int M, int N, int K,
               unsigned short* __restrict__ qd,         // MODE 0: [B,H,T,D]
               unsigned short* __restrict__ kd,         // MODE 0: [B,H,T,D]
               unsigned short* __restrict__ vtd,        // MODE 0: [B,H,D,T]
               float* __restrict__ outd)                // MODE 1: [M][N]
{
    // stride 72 halves = 144 B = 9*16 B: b128-aligned, odd multiple of 16B
    __shared__ __attribute__((aligned(16))) unsigned short As[2][128][72];
    __shared__ __attribute__((aligned(16))) unsigned short Bs[2][128][72];

    const int tid  = threadIdx.x;
    const int lane = tid & 31;
    const int w    = tid >> 5;
    const int wm   = w & 3;          // M quadrant (32 rows)
    const int wn   = w >> 2;         // N half (64 cols)
    const int h2   = lane >> 4;      // lane half
    const int ln   = lane & 15;

    const int m0 = blockIdx.y * 128;
    const int n0 = blockIdx.x * 128;

    f32x8 acc[2][4];
    for (int ms = 0; ms < 2; ++ms)
        for (int ns = 0; ns < 4; ++ns)
            acc[ms][ns] = (f32x8)0.0f;

    float biasn[4];
    for (int ns = 0; ns < 4; ++ns)
        biasn[ns] = bias[n0 + wn * 64 + ns * 16 + ln];

    // Issue one 128x64 stage of A and Bt as async DMA into LDS buffer `buf`.
    auto issue_stage = [&](int k0, int buf) {
        for (int i = 0; i < 4; ++i) {
            int idx = tid + i * 256;           // 0..1023
            int row = idx >> 3, ch = idx & 7;  // 8 chunks of 8 halves per row
            async_ld_b128(A  + (size_t)(m0 + row) * K + k0 + ch * 8,
                          &As[buf][row][ch * 8]);
            async_ld_b128(Bt + (size_t)(n0 + row) * K + k0 + ch * 8,
                          &Bs[buf][row][ch * 8]);
        }
    };

    issue_stage(0, 0);
    async_wait0();
    __syncthreads();

    const int nstages = K >> 6;                // K / 64
    for (int st = 0; st < nstages; ++st) {
        const int buf = st & 1;
        if (st + 1 < nstages) issue_stage((st + 1) << 6, buf ^ 1);

        // ---- compute current stage: 2 K-steps x 8 WMMAs ----
        for (int j = 0; j < 2; ++j) {
            FragB af[2], bfr[4];
            for (int ms = 0; ms < 2; ++ms) {
                const unsigned short* p = &As[buf][wm * 32 + ms * 16 + ln][32 * j + h2 * 8];
                af[ms].q[0] = *(const u32x4*)p;
                af[ms].q[1] = *(const u32x4*)(p + 16);
            }
            for (int ns = 0; ns < 4; ++ns) {
                const unsigned short* p = &Bs[buf][wn * 64 + ns * 16 + ln][32 * j + h2 * 8];
                bfr[ns].q[0] = *(const u32x4*)p;
                bfr[ns].q[1] = *(const u32x4*)(p + 16);
            }
            for (int ms = 0; ms < 2; ++ms)
                for (int ns = 0; ns < 4; ++ns)
                    acc[ms][ns] = __builtin_amdgcn_wmma_f32_16x16x32_bf16(
                        false, af[ms].v, false, bfr[ns].v,
                        (short)0, acc[ms][ns], false, false);
        }

        async_wait0();      // next stage's DMA landed (ASYNCcnt, not DScnt)
        __syncthreads();    // all waves done with `buf` + see new stage
    }

    // ---- epilogue ----
    for (int ms = 0; ms < 2; ++ms)
        for (int ns = 0; ns < 4; ++ns)
            for (int r = 0; r < 8; ++r) {
                int m = m0 + wm * 32 + ms * 16 + r + h2 * 8;   // C layout: M = vgpr + 8*half
                int n = n0 + wn * 64 + ns * 16 + ln;           // N = lane%16
                float val = acc[ms][ns][r] + biasn[ns];
                if (MODE == 0) {
                    int b = m >> 11, t = m & 2047;             // T = 2048
                    int which = n >> 10, e = n & 1023;         // q/k/v split
                    int hh = e >> 6, dd = e & 63;
                    size_t head = (size_t)b * HEADS + hh;
                    if (which == 0) {
                        val *= 0.125f;                         // 1/sqrt(64) folded into q
                        qd[((head * T_TOT + t) << 6) + dd] = f32_to_bf16(val);
                    } else if (which == 1) {
                        kd[((head * T_TOT + t) << 6) + dd] = f32_to_bf16(val);
                    } else {
                        vtd[((head * HDIM + dd) << 11) + t] = f32_to_bf16(val);
                    }
                } else {
                    outd[(size_t)m * N + n] = val;
                }
            }
}

// ---------------------------------------------------------------------------
// Flash attention: one block = 64 query rows of one (b,h); 4 waves x 16 rows.
// K / V^T tiles arrive via async DMA to LDS; Q stays in registers.
// ---------------------------------------------------------------------------
__global__ __launch_bounds__(128)
void attn_kernel(const unsigned short* __restrict__ qd,   // [B,H,T,D] (pre-scaled)
                 const unsigned short* __restrict__ kd,   // [B,H,T,D]
                 const unsigned short* __restrict__ vtd,  // [B,H,D,T]
                 unsigned short* __restrict__ yb)         // [B,T,E] bf16
{
    __shared__ __attribute__((aligned(16))) unsigned short Kt[32][72];      // [kk][d]
    __shared__ __attribute__((aligned(16))) unsigned short Vt[64][40];      // [d][kk]
    __shared__ __attribute__((aligned(16))) unsigned short Ps[4][16][40];   // per-wave P relayout

    const int tid  = threadIdx.x;
    const int lane = tid & 31;
    const int w    = tid >> 5;
    const int h2   = lane >> 4;
    const int ln   = lane & 15;

    const int bh = blockIdx.y;
    const int b  = bh >> 4, h = bh & 15;
    const int q0 = blockIdx.x * 64;
    const int qr = q0 + w * 16 + ln;

    const size_t head_off = (size_t)(b * HEADS + h) * T_TOT * HDIM; // 131072
    const unsigned short* kbase  = kd  + head_off;
    const unsigned short* vtbase = vtd + head_off;

    // Q fragments (16 rows x 64) held in registers for the whole loop
    const unsigned short* qrow = qd + head_off + ((size_t)qr << 6);
    FragB qa[2];
    for (int j = 0; j < 2; ++j) {
        qa[j].q[0] = *(const u32x4*)(qrow + 32 * j + 8 * h2);
        qa[j].q[1] = *(const u32x4*)(qrow + 32 * j + 16 + 8 * h2);
    }

    f32x8 yacc[4];
    for (int dt = 0; dt < 4; ++dt) yacc[dt] = (f32x8)0.0f;
    float mrow[8], lrow[8];
    for (int r = 0; r < 8; ++r) { mrow[r] = -3.0e38f; lrow[r] = 0.0f; }

    const bool causal = (q0 >= SEG);
    const int  limit  = causal ? (q0 + 64) : SEG;

    for (int kk0 = 0; kk0 < limit; kk0 += 32) {
        // ---- async DMA: K tile [32][64] and V^T tile [64][32] into LDS ----
        for (int i = 0; i < 2; ++i) {
            int idx = tid + i * 128;           // 0..255
            int row = idx >> 3, ch = idx & 7;
            async_ld_b128(kbase + ((size_t)(kk0 + row) << 6) + ch * 8,
                          &Kt[row][ch * 8]);
        }
        for (int i = 0; i < 2; ++i) {
            int idx = tid + i * 128;
            int row = idx >> 2, ch = idx & 3;
            async_ld_b128(vtbase + ((size_t)row << 11) + kk0 + ch * 8,
                          &Vt[row][ch * 8]);
        }
        async_wait0();
        __syncthreads();

        // ---- S = Q * K^T : 2 n-subtiles x 2 K-steps = 4 WMMA ----
        f32x8 s[2];
        for (int nt = 0; nt < 2; ++nt) {
            f32x8 a = (f32x8)0.0f;
            for (int j = 0; j < 2; ++j) {
                FragB kb;
                const unsigned short* p = &Kt[nt * 16 + ln][32 * j + 8 * h2];
                kb.q[0] = *(const u32x4*)p;
                kb.q[1] = *(const u32x4*)(p + 16);
                a = __builtin_amdgcn_wmma_f32_16x16x32_bf16(
                        false, qa[j].v, false, kb.v, (short)0, a, false, false);
            }
            s[nt] = a;
        }

        // ---- causal mask (elementwise, C layout) ----
        if (causal) {
            for (int nt = 0; nt < 2; ++nt)
                for (int r = 0; r < 8; ++r) {
                    int qg = q0 + w * 16 + r + h2 * 8;
                    int kg = kk0 + nt * 16 + ln;
                    if (kg > qg) s[nt][r] = -3.0e38f;
                }
        }

        // ---- online softmax: row lives in one 16-lane half ----
        float mnew[8], alpha[8];
        for (int r = 0; r < 8; ++r) {
            float v = fmaxf(s[0][r], s[1][r]);
            for (int msk = 1; msk < 16; msk <<= 1)
                v = fmaxf(v, __shfl_xor(v, msk, 32));
            mnew[r]  = fmaxf(mrow[r], v);
            alpha[r] = __expf(mrow[r] - mnew[r]);
            mrow[r]  = mnew[r];
        }
        for (int r = 0; r < 8; ++r) {
            float p0 = __expf(s[0][r] - mnew[r]);
            float p1 = __expf(s[1][r] - mnew[r]);
            s[0][r] = p0; s[1][r] = p1;
            float v = p0 + p1;
            for (int msk = 1; msk < 16; msk <<= 1)
                v += __shfl_xor(v, msk, 32);
            lrow[r] = lrow[r] * alpha[r] + v;
        }
        for (int dt = 0; dt < 4; ++dt)
            for (int r = 0; r < 8; ++r)
                yacc[dt][r] *= alpha[r];

        // ---- relayout P (C frag -> A frag) through wave-private LDS ----
        for (int nt = 0; nt < 2; ++nt)
            for (int r = 0; r < 8; ++r)
                Ps[w][r + h2 * 8][nt * 16 + ln] = f32_to_bf16(s[nt][r]);

        FragB pa;
        {
            const unsigned short* p = &Ps[w][ln][8 * h2];
            pa.q[0] = *(const u32x4*)p;
            pa.q[1] = *(const u32x4*)(p + 16);
        }

        // ---- y += P * V : 4 WMMA over d-subtiles ----
        for (int dt = 0; dt < 4; ++dt) {
            FragB vb;
            const unsigned short* p = &Vt[dt * 16 + ln][8 * h2];
            vb.q[0] = *(const u32x4*)p;
            vb.q[1] = *(const u32x4*)(p + 16);
            yacc[dt] = __builtin_amdgcn_wmma_f32_16x16x32_bf16(
                           false, pa.v, false, vb.v, (short)0, yacc[dt], false, false);
        }
        __syncthreads();
    }

    // ---- finalize: y /= l, store bf16 into [B,T,E] for the proj GEMM ----
    for (int r = 0; r < 8; ++r) {
        float inv = 1.0f / lrow[r];
        int qg = q0 + w * 16 + r + h2 * 8;
        for (int dt = 0; dt < 4; ++dt) {
            int e = h * HDIM + dt * 16 + ln;
            yb[((size_t)b * T_TOT + qg) * EMB + e] = f32_to_bf16(yacc[dt][r] * inv);
        }
    }
}

// ---------------------------------------------------------------------------
// Launch
// ---------------------------------------------------------------------------
extern "C" void kernel_launch(void* const* d_in, const int* in_sizes, int n_in,
                              void* d_out, int out_size, void* d_ws, size_t ws_size,
                              hipStream_t stream) {
    (void)in_sizes; (void)n_in; (void)out_size; (void)ws_size;

    const float* x     = (const float*)d_in[0];   // [4,2048,1024]
    const float* Wqkv  = (const float*)d_in[1];   // [1024,3072]
    const float* bqkv  = (const float*)d_in[2];   // [3072]
    const float* Wproj = (const float*)d_in[3];   // [1024,1024]
    const float* bproj = (const float*)d_in[4];   // [1024]
    float*       out   = (float*)d_out;           // [4,2048,1024]

    const int M  = BATCH * T_TOT;   // 8192
    const int K  = EMB;             // 1024
    const int N1 = 3 * EMB;         // 3072

    char* ws = (char*)d_ws;
    unsigned short* xb     = (unsigned short*)ws; ws += (size_t)M * K * 2;        // 16 MB
    unsigned short* WqkvT  = (unsigned short*)ws; ws += (size_t)N1 * K * 2;       // 6 MB
    unsigned short* WprojT = (unsigned short*)ws; ws += (size_t)K * K * 2;        // 2 MB
    unsigned short* qb     = (unsigned short*)ws; ws += (size_t)M * K * 2;        // 16 MB
    unsigned short* kb     = (unsigned short*)ws; ws += (size_t)M * K * 2;        // 16 MB
    unsigned short* vtb    = (unsigned short*)ws; ws += (size_t)M * K * 2;        // 16 MB
    unsigned short* yb     = (unsigned short*)ws;                                  // 16 MB

    // 1) precision conversion / weight transposes
    {
        int n = M * K;                                              // 8388608
        cvt_f32_bf16<<<(n + 255) / 256, 256, 0, stream>>>(x, xb, n);
    }
    {
        int n = K * N1;                                             // 3145728
        transpose_f32_bf16<<<(n + 255) / 256, 256, 0, stream>>>(Wqkv, WqkvT, K, N1);
    }
    {
        int n = K * K;                                              // 1048576
        transpose_f32_bf16<<<(n + 255) / 256, 256, 0, stream>>>(Wproj, WprojT, K, K);
    }

    // 2) fused QKV GEMM with scatter to q / k / v^T head layouts
    gemm_bf16<0><<<dim3(N1 / 128, M / 128), 256, 0, stream>>>(
        xb, WqkvT, bqkv, M, N1, K, qb, kb, vtb, nullptr);

    // 3) flash attention (prefix block + causal)
    attn_kernel<<<dim3(T_TOT / 64, BATCH * HEADS), 128, 0, stream>>>(qb, kb, vtb, yb);

    // 4) output projection (f32 out + bias)
    gemm_bf16<1><<<dim3(EMB / 128, M / 128), 256, 0, stream>>>(
        yb, WprojT, bproj, M, EMB, K, nullptr, nullptr, nullptr, out);
}
